// AlignableCaslAgent_29051158790177
// MI455X (gfx1250) — matrix operand order repaired and checked by reading, not compiled, not run on hardware
//
#include <hip/hip_runtime.h>
#include <cstddef>

// ---------------------------------------------------------------------------
// MI455X (gfx1250, wave32) implementation of the CASL agent forward step.
// GEMM stages: v_wmma_f32_16x16x32_f16 (f16 in, f32 acc).
// Weight/activation tiles staged via GLOBAL_LOAD_ASYNC_TO_LDS_B128 (ASYNCcnt),
// double-buffered in the FC/LSTM kernels.
// ---------------------------------------------------------------------------

typedef __attribute__((ext_vector_type(16))) _Float16 v16h;
typedef __attribute__((ext_vector_type(8)))  float    v8f;

#define B_ENV 1024
#define FEAT  512
#define HSZ   128

// ---------------- workspace layout (units: _Float16 elements) --------------
static constexpr size_t OFF_W1   = 0;                          // [2][32][64]
static constexpr size_t OFF_W2   = OFF_W1   + 2ull*32*64;      // [2][64][512]
static constexpr size_t OFF_W3   = OFF_W2   + 2ull*64*512;     // [2][64][576]
static constexpr size_t OFF_WL   = OFF_W3   + 2ull*64*576;     // [2][512][3136]
static constexpr size_t OFF_WCAT = OFF_WL   + 2ull*512*3136;   // [512][640]
static constexpr size_t OFF_A1   = OFF_WCAT + 512ull*640;      // [2][B][32][400]
static constexpr size_t OFF_A2   = OFF_A1   + 2ull*B_ENV*32*400; // [2][B][64][81]
static constexpr size_t OFF_A3   = OFF_A2   + 2ull*B_ENV*64*81;  // [2][B][3136]
static constexpr size_t OFF_AF   = OFF_A3   + 2ull*B_ENV*3136;   // [B][640]

// ---------------- CDNA5 async global->LDS helpers --------------------------
__device__ inline void async_b128(unsigned lds_off, const void* gptr) {
  // Copies 16B global -> LDS per lane, tracked by ASYNCcnt, bypasses VGPRs.
  asm volatile("global_load_async_to_lds_b128 %0, %1, off"
               :: "v"(lds_off), "v"((unsigned long long)gptr)
               : "memory");
}

template <int N>
__device__ inline void wait_asynccnt() {
  asm volatile("s_wait_asynccnt %0" :: "i"(N) : "memory");
}

__device__ inline unsigned lds_addr(const void* p) {
  // gfx1250: generic pointer to LDS carries the LDS byte offset in addr[31:0].
  return (unsigned)(size_t)p;
}

// ---------------- WMMA fragment helpers (CDNA5 wave32 layouts) -------------
__device__ inline v16h load_a_frag(const _Float16* a, int lda) {
  // A tile: [16 rows (M)] x [32 cols (K)], row-major with stride lda.
  const int lane = threadIdx.x & 31;
  const int m = lane & 15, g = lane >> 4;
  v16h r;
#pragma unroll
  for (int h = 0; h < 16; ++h) {
    const int k = (h < 8) ? (h + g * 8) : (h + 8 + g * 8); // 16+(h-8)+8g
    r[h] = a[m * lda + k];
  }
  return r;
}

__device__ inline v16h load_b_frag(const _Float16* bt, int ldb) {
  // bt: weight tile stored [n][k] (row-major [N][K]); supplies B[k][n].
  const int lane = threadIdx.x & 31;
  const int n = lane & 15, g = lane >> 4;
  v16h r;
#pragma unroll
  for (int h = 0; h < 16; ++h) {
    const int k = (h < 8) ? (h + g * 8) : (h + 8 + g * 8);
    r[h] = bt[n * ldb + k];
  }
  return r;
}

__device__ inline v8f wmma16(v16h a, v16h b, v8f c) {
  return __builtin_amdgcn_wmma_f32_16x16x32_f16(
      /*neg_a=*/false, a, /*neg_b=*/false, b,
      /*c_mod=*/(short)0, c, /*reuse_a=*/false, /*reuse_b=*/false);
}

// ---------------- weight conversion ----------------------------------------
__global__ void k_cvt(const float* __restrict__ in, _Float16* __restrict__ out,
                      int n) {
  int i = blockIdx.x * 256 + threadIdx.x;
  if (i < n) out[i] = (_Float16)in[i];
}

__global__ void k_wcat(const float* __restrict__ wih,
                       const float* __restrict__ whh,
                       _Float16* __restrict__ out) {
  int i = blockIdx.x * 256 + threadIdx.x; // 512*640
  if (i < 512 * 640) {
    int n = i / 640, k = i - n * 640;
    float v = (k < 512) ? wih[n * 512 + k] : whh[n * 128 + (k - 512)];
    out[i] = (_Float16)v;
  }
}

// ---------------- implicit-GEMM conv (both towers, WMMA) -------------------
// Block: 256 threads = 8 waves. Tile: M=64 x N=COUT. K chunked by 32.
// A tile: im2col VALU gather -> LDS. B tile: async b128 global -> LDS.
template <bool IS_X, int CIN, int IH, int IW, int KH, int KW, int STR, int OH,
          int OW, int COUT>
__global__ void __launch_bounds__(256)
k_conv_gemm(const float* __restrict__ xin,      // used when IS_X (B,2,84,84)
            const _Float16* __restrict__ ain,   // otherwise [2][B][CIN][IH*IW]
            const _Float16* __restrict__ wts,   // [2][COUT][K_TOT]
            const float* __restrict__ bias0, const float* __restrict__ bias1,
            _Float16* __restrict__ aout) {      // [2][B][COUT][OH*OW]
  constexpr int K_TOT = CIN * KH * KW;
  constexpr int P = OH * OW;
  constexpr int NT = COUT / 16;   // N sub-tiles
  constexpr int ACC = COUT / 32;  // accumulators per wave (1 or 2)

  const int t = blockIdx.y;            // tower
  const int rowBase = blockIdx.x * 64; // im2col row
  const int tid = threadIdx.x;
  const int wave = tid >> 5, lane = tid & 31;

  __shared__ _Float16 ldsA[64][34];
  __shared__ _Float16 ldsB[COUT][40]; // 80B row stride: 16B-aligned b128 blocks

  v8f acc[ACC] = {};
  const int slot0 = wave * ACC;
  const int m_sub = slot0 / NT;
  const _Float16* wrow = wts + (size_t)t * COUT * K_TOT;

  for (int k0 = 0; k0 < K_TOT; k0 += 32) {
    // --- B tile: COUT x 32 halves via async b128 (COUT*4 blocks of 16B) ----
    if (tid < COUT * 4) {
      const int n = tid >> 2, kb = tid & 3;
      async_b128(lds_addr(&ldsB[n][kb * 8]),
                 wrow + (size_t)n * K_TOT + k0 + kb * 8);
    }
    // --- A tile (im2col gather) : 64x32 f16 --------------------------------
#pragma unroll
    for (int e = 0; e < 8; ++e) {
      const int flat = tid + e * 256; // 2048 elements
      const int m = flat >> 5, kk = flat & 31;
      const int r = rowBase + m;
      const int b = r / P, p = r - b * P;
      const int oy = p / OW, ox = p - oy * OW;
      const int k = k0 + kk;
      const int c = k / (KH * KW);
      const int rem = k - c * (KH * KW);
      const int ky = rem / KW, kx = rem - ky * KW;
      const int iy = oy * STR + ky, ix = ox * STR + kx;
      _Float16 v;
      if constexpr (IS_X) {
        v = (_Float16)xin[(((size_t)b * 2 + t) * IH + iy) * IW + ix];
      } else {
        v = ain[((((size_t)t * B_ENV + b) * CIN + c) * IH + iy) * IW + ix];
      }
      ldsA[m][kk] = v;
    }
    wait_asynccnt<0>();
    __syncthreads();

    const v16h afrag = load_a_frag(&ldsA[m_sub * 16][0], 34);
#pragma unroll
    for (int s = 0; s < ACC; ++s) {
      const int n_sub = (slot0 + s) % NT;
      const v16h bfrag = load_b_frag(&ldsB[n_sub * 16][0], 40);
      acc[s] = wmma16(afrag, bfrag, acc[s]);
    }
    __syncthreads();
  }

  // --- epilogue: bias + ReLU, store f16 NCHW -------------------------------
  const float* bias = (t == 0) ? bias0 : bias1;
  const int g = lane >> 4, nl = lane & 15;
#pragma unroll
  for (int s = 0; s < ACC; ++s) {
    const int n = ((slot0 + s) % NT) * 16 + nl;
    const float bn = bias[n];
#pragma unroll
    for (int rr = 0; rr < 8; ++rr) {
      const int r = rowBase + m_sub * 16 + rr + g * 8;
      const int b = r / P, p = r - b * P;
      float v = acc[s][rr] + bn;
      v = v > 0.f ? v : 0.f;
      aout[(((size_t)t * B_ENV + b) * COUT + n) * P + p] = (_Float16)v;
    }
  }
}

// ---------------- full-row GEMM building blocks (16 rows x 512 cols) -------
// Double-buffered: async-stage chunk i+1 while WMMAing chunk i.
// Per-wave async ops/chunk: 8 (B tile) + 1 for waves 0,1 (A tile).
template <int K_TOT>
__device__ inline void stage_chunk(const _Float16* arow, const _Float16* wrow,
                                   _Float16 (&ldsA)[2][16][40],
                                   _Float16 (&ldsB)[2][512][40], int k0,
                                   int buf) {
  const int tid = threadIdx.x;
#pragma unroll
  for (int e = 0; e < 8; ++e) { // B: 512 rows x 4 blocks = 2048 b128 blocks
    const int blk = tid + e * 256;
    const int n = blk >> 2, kb = blk & 3;
    async_b128(lds_addr(&ldsB[buf][n][kb * 8]),
               wrow + (size_t)n * K_TOT + k0 + kb * 8);
  }
  if (tid < 64) { // A: 16 rows x 4 blocks (waves 0,1 only -> wave-uniform)
    const int m = tid >> 2, kb = tid & 3;
    async_b128(lds_addr(&ldsA[buf][m][kb * 8]),
               arow + (size_t)m * K_TOT + k0 + kb * 8);
  }
}

__device__ inline void compute_chunk(const _Float16 (&ldsA)[2][16][40],
                                     const _Float16 (&ldsB)[2][512][40],
                                     int buf, int wave, v8f acc[4]) {
  const v16h afrag = load_a_frag(&ldsA[buf][0][0], 40);
#pragma unroll
  for (int s = 0; s < 4; ++s) {
    const v16h bfrag = load_b_frag(&ldsB[buf][(wave * 4 + s) * 16][0], 40);
    acc[s] = wmma16(afrag, bfrag, acc[s]);
  }
}

template <int K_TOT>
__device__ inline void gemm_pipeline(const _Float16* arow,
                                     const _Float16* wrow,
                                     _Float16 (&ldsA)[2][16][40],
                                     _Float16 (&ldsB)[2][512][40], int wave,
                                     v8f acc[4]) {
  constexpr int NCH = K_TOT / 32;
  stage_chunk<K_TOT>(arow, wrow, ldsA, ldsB, 0, 0);
  for (int i = 0; i < NCH; ++i) {
    if (i + 1 < NCH) {
      stage_chunk<K_TOT>(arow, wrow, ldsA, ldsB, (i + 1) * 32, (i + 1) & 1);
      // Async loads retire in order: waiting down to the ops just issued
      // guarantees chunk i's copies have landed in LDS.
      if (wave < 2) wait_asynccnt<9>();
      else          wait_asynccnt<8>();
    } else {
      wait_asynccnt<0>();
    }
    __syncthreads(); // all waves' chunk-i data visible
    compute_chunk(ldsA, ldsB, i & 1, wave, acc);
    __syncthreads(); // safe to overwrite buf (i&1) two chunks later
  }
}

// ---------------- FC 3136->512 + ReLU + LayerNorm --------------------------
__global__ void __launch_bounds__(256)
k_linear_ln(const _Float16* __restrict__ act3, // [2][B][3136]
            const _Float16* __restrict__ wl,   // [2][512][3136]
            const float* __restrict__ bl0, const float* __restrict__ bl1,
            const float* __restrict__ g0, const float* __restrict__ lb0,
            const float* __restrict__ g1, const float* __restrict__ lb1,
            float* __restrict__ outv, float* __restrict__ outa) {
  constexpr int K_TOT = 3136;
  const int t = blockIdx.y;
  const int rowBase = blockIdx.x * 16;
  const int tid = threadIdx.x, wave = tid >> 5, lane = tid & 31;

  __shared__ _Float16 ldsA[2][16][40];
  __shared__ _Float16 ldsB[2][512][40];
  __shared__ float ldsO[16][512];

  v8f acc[4] = {};
  const _Float16* arow = act3 + ((size_t)t * B_ENV + rowBase) * K_TOT;
  const _Float16* wrow = wl + (size_t)t * 512 * K_TOT;

  gemm_pipeline<K_TOT>(arow, wrow, ldsA, ldsB, wave, acc);

  { // scatter accumulators to ldsO
    const int g = lane >> 4, nl = lane & 15;
#pragma unroll
    for (int s = 0; s < 4; ++s) {
      const int n = (wave * 4 + s) * 16 + nl;
#pragma unroll
      for (int rr = 0; rr < 8; ++rr) ldsO[rr + g * 8][n] = acc[s][rr];
    }
  }
  __syncthreads();

  const float* bl = t ? bl1 : bl0;
  const float* lg = t ? g1 : g0;
  const float* lb = t ? lb1 : lb0;
  float* out = t ? outa : outv;
#pragma unroll
  for (int e = 0; e < 2; ++e) { // 2 rows per wave
    const int row = wave * 2 + e;
    float sum = 0.f, sq = 0.f;
    for (int j = lane; j < 512; j += 32) {
      float v = ldsO[row][j] + bl[j];
      v = v > 0.f ? v : 0.f;
      ldsO[row][j] = v;
      sum += v;
      sq += v * v;
    }
#pragma unroll
    for (int off = 16; off; off >>= 1) {
      sum += __shfl_xor(sum, off, 32);
      sq += __shfl_xor(sq, off, 32);
    }
    const float mean = sum * (1.f / 512.f);
    const float var = sq * (1.f / 512.f) - mean * mean;
    const float rstd = rsqrtf(var + 1e-5f);
    const size_t ob = (size_t)(rowBase + row) * 512;
    for (int j = lane; j < 512; j += 32)
      out[ob + j] = (ldsO[row][j] - mean) * rstd * lg[j] + lb[j];
  }
}

// ---------------- attention gate + A-matrix build for LSTM -----------------
__global__ void __launch_bounds__(256)
k_attn(const float* __restrict__ vf, const float* __restrict__ af,
       const float* __restrict__ h0, const float* __restrict__ done,
       const float* __restrict__ avW, const float* __restrict__ avb,
       const float* __restrict__ aaW, const float* __restrict__ aab,
       const float* __restrict__ asW, const float* __restrict__ asb,
       const float* __restrict__ attW, const float* __restrict__ attb,
       _Float16* __restrict__ afull) { // [B][640] = [fused | masked-h]
  const int wave = threadIdx.x >> 5, lane = threadIdx.x & 31;
  const int b = blockIdx.x * 8 + wave;
  const float* vrow = vf + (size_t)b * 512;
  const float* arow = af + (size_t)b * 512;
  const float* hrow = h0 + (size_t)b * 128;

  // lane j computes act[j], j = 0..31 (attention uses UNmasked h0)
  float s = avb[lane] + aab[lane] + asb[lane];
  for (int k = 0; k < 512; ++k)
    s += vrow[k] * avW[lane * 512 + k] + arow[k] * aaW[lane * 512 + k];
  for (int k = 0; k < 128; ++k) s += hrow[k] * asW[lane * 128 + k];
  const float a = tanhf(s);

  float p0 = a * attW[lane], p1 = a * attW[32 + lane];
#pragma unroll
  for (int off = 16; off; off >>= 1) {
    p0 += __shfl_xor(p0, off, 32);
    p1 += __shfl_xor(p1, off, 32);
  }
  const float l0 = p0 + attb[0], l1 = p1 + attb[1];
  const float mx = fmaxf(l0, l1);
  const float e0 = __expf(l0 - mx), e1 = __expf(l1 - mx);
  const float w0 = e0 / (e0 + e1), w1 = 1.f - w0;

  _Float16* dst = afull + (size_t)b * 640;
  for (int k = lane; k < 512; k += 32)
    dst[k] = (_Float16)(w0 * vrow[k] + w1 * arow[k]);
  const float m = 1.f - done[b];
  for (int k = lane; k < 128; k += 32)
    dst[512 + k] = (_Float16)(m * hrow[k]);
}

// ---------------- LSTM step: gates GEMM (K=640) + elementwise --------------
__global__ void __launch_bounds__(256)
k_lstm(const _Float16* __restrict__ afull, // [B][640]
       const _Float16* __restrict__ wcat,  // [512][640] = [Wih | Whh]
       const float* __restrict__ bih, const float* __restrict__ bhh,
       const float* __restrict__ c0, const float* __restrict__ done,
       float* __restrict__ hout, float* __restrict__ cout) {
  constexpr int K_TOT = 640;
  const int rowBase = blockIdx.x * 16;
  const int tid = threadIdx.x, wave = tid >> 5, lane = tid & 31;

  __shared__ _Float16 ldsA[2][16][40];
  __shared__ _Float16 ldsB[2][512][40];
  __shared__ float ldsO[16][512];

  v8f acc[4] = {};
  const _Float16* arow = afull + (size_t)rowBase * K_TOT;

  gemm_pipeline<K_TOT>(arow, wcat, ldsA, ldsB, wave, acc);

  {
    const int g = lane >> 4, nl = lane & 15;
#pragma unroll
    for (int s = 0; s < 4; ++s) {
      const int n = (wave * 4 + s) * 16 + nl;
#pragma unroll
      for (int rr = 0; rr < 8; ++rr) ldsO[rr + g * 8][n] = acc[s][rr];
    }
  }
  __syncthreads();

  // elementwise LSTM: 16 rows x 128 hidden = 2048 items
#pragma unroll
  for (int e = 0; e < 8; ++e) {
    const int flat = tid + e * 256;
    const int row = flat >> 7, j = flat & 127;
    const int b = rowBase + row;
    const float gi = ldsO[row][j] + bih[j] + bhh[j];
    const float gf = ldsO[row][128 + j] + bih[128 + j] + bhh[128 + j];
    const float gg = ldsO[row][256 + j] + bih[256 + j] + bhh[256 + j];
    const float go = ldsO[row][384 + j] + bih[384 + j] + bhh[384 + j];
    const float m = 1.f - done[b];
    const float c = m * c0[(size_t)b * 128 + j];
    const float si = 1.f / (1.f + __expf(-gi));
    const float sf = 1.f / (1.f + __expf(-gf));
    const float so = 1.f / (1.f + __expf(-go));
    const float cn = sf * c + si * tanhf(gg);
    const float hn = so * tanhf(cn);
    hout[(size_t)b * 128 + j] = hn;
    cout[(size_t)b * 128 + j] = cn;
  }
}

// ---------------------------------------------------------------------------
extern "C" void kernel_launch(void* const* d_in, const int* in_sizes, int n_in,
                              void* d_out, int out_size, void* d_ws,
                              size_t ws_size, hipStream_t stream) {
  (void)in_sizes; (void)n_in; (void)out_size; (void)ws_size;
  _Float16* wsh = (_Float16*)d_ws;
  float* out = (float*)d_out;
  float* h_new = out;                 // [B][128]
  float* c_new = out + 131072;        // [B][128]
  float* vf = out + 262144;           // [B][512]
  float* af = out + 786432;           // [B][512]

  auto cvt = [&](int idx, size_t dstOff, int n) {
    k_cvt<<<(n + 255) / 256, 256, 0, stream>>>((const float*)d_in[idx],
                                               wsh + dstOff, n);
  };
  // weight conversions (fp32 -> f16)
  cvt(4, OFF_W1, 32 * 64);                  // vW1
  cvt(12, OFF_W1 + 32 * 64, 32 * 64);       // aW1
  cvt(6, OFF_W2, 64 * 512);                 // vW2
  cvt(14, OFF_W2 + 64 * 512, 64 * 512);     // aW2
  cvt(8, OFF_W3, 64 * 576);                 // vW3
  cvt(16, OFF_W3 + 64 * 576, 64 * 576);     // aW3
  cvt(10, OFF_WL, 512 * 3136);              // vWl
  cvt(18, OFF_WL + 512 * 3136, 512 * 3136); // aWl
  k_wcat<<<(512 * 640 + 255) / 256, 256, 0, stream>>>(
      (const float*)d_in[32], (const float*)d_in[33], wsh + OFF_WCAT);

  // conv towers (grid.y = tower)
  k_conv_gemm<true, 1, 84, 84, 8, 8, 4, 20, 20, 32>
      <<<dim3(B_ENV * 400 / 64, 2), 256, 0, stream>>>(
          (const float*)d_in[0], nullptr, wsh + OFF_W1,
          (const float*)d_in[5], (const float*)d_in[13], wsh + OFF_A1);
  k_conv_gemm<false, 32, 20, 20, 4, 4, 2, 9, 9, 64>
      <<<dim3(B_ENV * 81 / 64, 2), 256, 0, stream>>>(
          nullptr, wsh + OFF_A1, wsh + OFF_W2,
          (const float*)d_in[7], (const float*)d_in[15], wsh + OFF_A2);
  k_conv_gemm<false, 64, 9, 9, 3, 3, 1, 7, 7, 64>
      <<<dim3(B_ENV * 49 / 64, 2), 256, 0, stream>>>(
          nullptr, wsh + OFF_A2, wsh + OFF_W3,
          (const float*)d_in[9], (const float*)d_in[17], wsh + OFF_A3);

  // FC + ReLU + LayerNorm -> vf/af (f32, straight into d_out)
  k_linear_ln<<<dim3(B_ENV / 16, 2), 256, 0, stream>>>(
      wsh + OFF_A3, wsh + OFF_WL, (const float*)d_in[11],
      (const float*)d_in[19], (const float*)d_in[20], (const float*)d_in[21],
      (const float*)d_in[22], (const float*)d_in[23], vf, af);

  // attention gate -> fused/masked-h f16 A-matrix
  k_attn<<<B_ENV / 8, 256, 0, stream>>>(
      vf, af, (const float*)d_in[2], (const float*)d_in[1],
      (const float*)d_in[24], (const float*)d_in[25], (const float*)d_in[26],
      (const float*)d_in[27], (const float*)d_in[28], (const float*)d_in[29],
      (const float*)d_in[30], (const float*)d_in[31], wsh + OFF_AF);

  // LSTM step
  k_lstm<<<B_ENV / 16, 256, 0, stream>>>(
      wsh + OFF_AF, wsh + OFF_WCAT, (const float*)d_in[34],
      (const float*)d_in[35], (const float*)d_in[3], (const float*)d_in[1],
      h_new, c_new);
}
